// HopfieldMemoryLayer_7430293422763
// MI455X (gfx1250) — matrix operand order repaired and checked by reading, not compiled, not run on hardware
//
#include <hip/hip_runtime.h>
#include <math.h>

// ---------------- shapes (fixed by the reference) ----------------
#define B_    4
#define S_    1024
#define D_    2048
#define M_    768
#define N_    4096
#define H_    4
#define DH_   256
#define A_    1024           // H_*DH_
#define BETA_     0.5f
#define CONV_EPS_ 1e-4f
#define LN_EPS_   1e-5f

// padded LDS row strides (ushorts); both are 16B multiples for b128 access
#define LDK_  264            // 32 x 264 ushorts  (k chunk, row-major [n][d])
#define LDKT_ 40             // 256 x 40 ushorts  (k^T chunk, [d][n])

typedef __attribute__((ext_vector_type(16))) __bf16 v16bf;
typedef __attribute__((ext_vector_type(8)))  float  v8f;

#define DEV_INLINE __device__ __forceinline__

// ---------------- small helpers ----------------
DEV_INLINE unsigned short f2bf(float f) {
  union { float f; unsigned int u; } x; x.f = f;
  unsigned int u = x.u;
  unsigned int r = u + 0x7FFFu + ((u >> 16) & 1u);
  if ((u & 0x7F800000u) == 0x7F800000u) r = u;   // inf/nan passthrough
  return (unsigned short)(r >> 16);
}

DEV_INLINE v8f wmma_bf16(v16bf a, v16bf b, v8f c) {
  // D = A(16x32 bf16) * B(32x16 bf16) + C(16x16 f32)
  return __builtin_amdgcn_wmma_f32_16x16x32_bf16(
      /*neg_a=*/false, a, /*neg_b=*/false, b,
      /*c_mod=*/(short)0, c, /*reuse_a=*/false, /*reuse_b=*/false);
}

// A-matrix fragment (16 rows x 32 K), X row-major (ld in ushorts).
// ISA layout: lane L: m = L%16; lanes 0-15 hold K 0..7 & 16..23, lanes 16-31 hold K 8..15 & 24..31.
DEV_INLINE v16bf load_frag_a(const unsigned short* X, int ldx, int row0, int k0) {
  int l = threadIdx.x & 31;
  int m = l & 15, h = l >> 4;
  const unsigned short* p = X + (size_t)(row0 + m) * ldx + k0 + 8 * h;
  union { v16bf v; uint4 q[2]; } u;
  u.q[0] = *(const uint4*)(p);        // K = 8h .. 8h+7
  u.q[1] = *(const uint4*)(p + 16);   // K = 16+8h .. 16+8h+7
  return u.v;
}

// B-matrix fragment (32 K x 16 cols) given B^T row-major: Bt[col][k] (ld in ushorts).
// ISA layout: lane L: col = L%16; lanes 0-15 hold K 0..7 & 16..23, lanes 16-31 hold K 8..15 & 24..31.
DEV_INLINE v16bf load_frag_b(const unsigned short* Bt, int ldb, int col0, int k0) {
  int l = threadIdx.x & 31;
  int n = l & 15, h = l >> 4;
  const unsigned short* p = Bt + (size_t)(col0 + n) * ldb + k0 + 8 * h;
  union { v16bf v; uint4 q[2]; } u;
  u.q[0] = *(const uint4*)(p);
  u.q[1] = *(const uint4*)(p + 16);
  return u.v;
}

// reduce across the 16 lanes of each half-wave (rows live per-half in C layout)
DEV_INLINE float half_reduce_max(float x) {
  for (int m = 1; m <= 8; m <<= 1) x = fmaxf(x, __shfl_xor(x, m, 32));
  return x;
}
DEV_INLINE float half_reduce_sum(float x) {
  for (int m = 1; m <= 8; m <<= 1) x += __shfl_xor(x, m, 32);
  return x;
}
DEV_INLINE float wave_reduce_sum(float x) {
  for (int m = 1; m <= 16; m <<= 1) x += __shfl_xor(x, m, 32);
  return x;
}

// ---------------- LayerNorm (fp32 in) -> bf16 out ----------------
__global__ void ln_bf16_kernel(const float* __restrict__ X,
                               const float* __restrict__ g,
                               const float* __restrict__ b,
                               unsigned short* __restrict__ Y, int C) {
  int row = blockIdx.x;
  const float* x = X + (size_t)row * C;
  float s = 0.f, s2 = 0.f;
  for (int i = threadIdx.x; i < C; i += blockDim.x) { float v = x[i]; s += v; s2 += v * v; }
  for (int m = 1; m < 32; m <<= 1) { s += __shfl_xor(s, m, 32); s2 += __shfl_xor(s2, m, 32); }
  __shared__ float shs[8], shs2[8];
  __shared__ float tot[2];
  int wid = threadIdx.x >> 5;
  if ((threadIdx.x & 31) == 0) { shs[wid] = s; shs2[wid] = s2; }
  __syncthreads();
  if (threadIdx.x == 0) {
    float a = 0.f, c2 = 0.f;
    int nw = blockDim.x >> 5;
    for (int i = 0; i < nw; ++i) { a += shs[i]; c2 += shs2[i]; }
    tot[0] = a; tot[1] = c2;
  }
  __syncthreads();
  float mu = tot[0] / (float)C;
  float var = tot[1] / (float)C - mu * mu;
  float r = rsqrtf(var + LN_EPS_);
  unsigned short* y = Y + (size_t)row * C;
  for (int i = threadIdx.x; i < C; i += blockDim.x)
    y[i] = f2bf((x[i] - mu) * r * g[i] + b[i]);
}

// ---------------- fp32 -> bf16 (flat) ----------------
__global__ void f32_to_bf16_kernel(const float* __restrict__ in,
                                   unsigned short* __restrict__ out, size_t n) {
  size_t t = (size_t)blockIdx.x * blockDim.x + threadIdx.x;
  if (t < n) out[t] = f2bf(in[t]);
}

// ---------------- fp32 (R,C) -> bf16 (C,R) transpose ----------------
__global__ void transpose_bf16_kernel(const float* __restrict__ in,
                                      unsigned short* __restrict__ out, int R, int C) {
  size_t t = (size_t)blockIdx.x * blockDim.x + threadIdx.x;
  size_t total = (size_t)R * C;
  if (t >= total) return;
  int c = (int)(t / R);
  int r = (int)(t % R);
  out[t] = f2bf(in[(size_t)r * C + c]);
}

// ---------------- bf16 GEMM: C(RxOut) = X(RxK) * W(OutxK)^T + bias ----------------
// one wave per block computes a 32x32 tile with 2x2 WMMA accumulators
__global__ void __launch_bounds__(32) gemm_bf16_kernel(
    const unsigned short* __restrict__ X,   // (R,K) bf16
    const unsigned short* __restrict__ W,   // (Out,K) bf16 (== B^T row-major)
    const float* __restrict__ bias,         // (Out) or null
    float* __restrict__ C,                  // (R,Out) fp32
    int R, int K, int Out) {
  int col0 = blockIdx.x * 32;
  int row0 = blockIdx.y * 32;
  v8f a00 = {0,0,0,0,0,0,0,0}, a01 = a00, a10 = a00, a11 = a00;
  for (int k0 = 0; k0 < K; k0 += 32) {
    v16bf xa0 = load_frag_a(X, K, row0, k0);
    v16bf xa1 = load_frag_a(X, K, row0 + 16, k0);
    v16bf wb0 = load_frag_b(W, K, col0, k0);
    v16bf wb1 = load_frag_b(W, K, col0 + 16, k0);
    a00 = wmma_bf16(xa0, wb0, a00);
    a01 = wmma_bf16(xa0, wb1, a01);
    a10 = wmma_bf16(xa1, wb0, a10);
    a11 = wmma_bf16(xa1, wb1, a11);
  }
  int l = threadIdx.x & 31, nn = l & 15, hh = l >> 4;
  float bi0 = bias ? bias[col0 + nn] : 0.f;
  float bi1 = bias ? bias[col0 + 16 + nn] : 0.f;
  for (int v = 0; v < 8; ++v) {
    int r0 = row0 + v + 8 * hh, r1 = r0 + 16;
    C[(size_t)r0 * Out + col0 + nn]        = a00[v] + bi0;
    C[(size_t)r0 * Out + col0 + 16 + nn]   = a01[v] + bi1;
    C[(size_t)r1 * Out + col0 + nn]        = a10[v] + bi0;
    C[(size_t)r1 * Out + col0 + 16 + nn]   = a11[v] + bi1;
  }
}

// ---------------- flash Hopfield attention step ----------------
// grid: (S/64, H, B), block = 128 threads = 4 waves (one WGP).
// Each wave owns one 16-query tile; the 4 waves cooperatively stage each
// 32-key chunk into LDS (both layouts) so k/kT are read from L2 once per block.
__global__ void __launch_bounds__(128) hopfield_attn_kernel(
    const unsigned short* __restrict__ xi_bf,  // (B*S, A) bf16
    const unsigned short* __restrict__ k_bf,   // (N, A)   bf16
    const unsigned short* __restrict__ bT_bf,  // (A, N)   bf16: kT (updates) or vT (final)
    const float* __restrict__ xi_old,          // (B*S, A) fp32
    float* __restrict__ xi_new,                // (B*S, A) fp32 (update steps)
    float* __restrict__ diff,                  // (B*H) fp32, pre-zeroed
    unsigned short* __restrict__ out_bf,       // (B*S, A) bf16 (final step)
    int is_final) {
  int h = blockIdx.y, b = blockIdx.z;
  int tid = threadIdx.x;
  int w = tid >> 5;                    // wave id 0..3
  int l = tid & 31, ln16 = l & 15, hh = l >> 4;
  int st = blockIdx.x * 4 + w;         // this wave's 16-query tile
  int rowBase = b * S_ + st * 16;

  __shared__ unsigned short ldsK[32 * LDK_];    // k chunk  [n][d], padded
  __shared__ unsigned short ldsKT[256 * LDKT_]; // kT chunk [d][n], padded
  __shared__ unsigned short pShm[4 * 16 * 32];  // per-wave prob tile staging
  unsigned short* myP = pShm + w * (16 * 32);

  // persistent xi A-fragments over the whole N loop (reused 128x)
  v16bf aX[8];
  for (int i = 0; i < 8; ++i)
    aX[i] = load_frag_a(xi_bf, A_, rowBase, h * DH_ + 32 * i);

  v8f acc[16];
  const v8f vzero = {0,0,0,0,0,0,0,0};
  for (int t = 0; t < 16; ++t) acc[t] = vzero;

  float mrow[8], lrow[8];
  for (int v = 0; v < 8; ++v) { mrow[v] = -1e30f; lrow[v] = 0.f; }

  const unsigned short* Bt = bT_bf + (size_t)h * DH_ * N_;  // rows d (256), cols N

  for (int n0 = 0; n0 < N_; n0 += 32) {
    // ---- cooperative staging of this key chunk into LDS ----
    __syncthreads();   // previous iteration's LDS reads complete
    // k chunk: 32 rows x 128 d = 16 uint4 per row
    for (int idx = tid; idx < 32 * 16; idx += 128) {
      int r = idx >> 4, c = (idx & 15) * 8;
      *(uint4*)&ldsK[r * LDK_ + c] =
          *(const uint4*)&k_bf[(size_t)(n0 + r) * A_ + h * DH_ + c];
    }
    // kT chunk: 256 rows x 32 n = 4 uint4 per row
    for (int idx = tid; idx < 256 * 4; idx += 128) {
      int r = idx >> 2, c = (idx & 3) * 8;
      *(uint4*)&ldsKT[r * LDKT_ + c] =
          *(const uint4*)&Bt[(size_t)r * N_ + n0 + c];
    }
    __syncthreads();   // staged data visible

    // ---- scores: two 16x16 tiles, accumulate over d=256 (B-frags from LDS) ----
    v8f s0 = vzero, s1 = vzero;
    for (int i = 0; i < 8; ++i) {
      v16bf b0 = load_frag_b(ldsK, LDK_, 0,  32 * i);
      v16bf b1 = load_frag_b(ldsK, LDK_, 16, 32 * i);
      s0 = wmma_bf16(aX[i], b0, s0);
      s1 = wmma_bf16(aX[i], b1, s1);
    }

    // ---- online softmax over the 32 new keys ----
    float p0[8], p1[8], scale[8];
    for (int v = 0; v < 8; ++v) {
      float x0 = BETA_ * s0[v], x1 = BETA_ * s1[v];
      float mx = half_reduce_max(fmaxf(x0, x1));
      float mnew = fmaxf(mrow[v], mx);
      float e0 = __expf(x0 - mnew), e1 = __expf(x1 - mnew);
      float rs = half_reduce_sum(e0 + e1);
      float corr = __expf(mrow[v] - mnew);
      lrow[v] = lrow[v] * corr + rs;
      mrow[v] = mnew;
      scale[v] = corr;
      p0[v] = e0; p1[v] = e1;
    }
    for (int t = 0; t < 16; ++t)
      for (int v = 0; v < 8; ++v) acc[t][v] *= scale[v];

    // ---- C-layout probs -> A-layout bf16 fragment via per-wave LDS region ----
    for (int v = 0; v < 8; ++v) {
      int r = (v + 8 * hh) * 32 + ln16;
      myP[r]      = f2bf(p0[v]);
      myP[r + 16] = f2bf(p1[v]);
    }
    __syncthreads();   // own writes ordered before read; uniform across block
    v16bf pA = load_frag_a(myP, 32, 0, 0);

    // ---- accumulate out += P(16x32) @ chunk(32 x 256) (B-frags from LDS kT) ----
    for (int t = 0; t < 16; ++t) {
      v16bf bv = load_frag_b(ldsKT, LDKT_, t * 16, 0);
      acc[t] = wmma_bf16(pA, bv, acc[t]);
    }
  }

  float inv[8];
  for (int v = 0; v < 8; ++v) inv[v] = 1.0f / lrow[v];

  if (!is_final) {
    float dsum = 0.f;
    for (int t = 0; t < 16; ++t) {
      for (int v = 0; v < 8; ++v) {
        int s = st * 16 + v + 8 * hh;
        int d = t * 16 + ln16;
        size_t addr = ((size_t)(b * S_ + s)) * A_ + h * DH_ + d;
        float xn = acc[t][v] * inv[v];
        float df = xn - xi_old[addr];
        dsum += df * df;
        xi_new[addr] = xn;
      }
    }
    dsum = wave_reduce_sum(dsum);
    if (l == 0) atomicAdd(&diff[b * H_ + h], dsum);
  } else {
    for (int t = 0; t < 16; ++t) {
      for (int v = 0; v < 8; ++v) {
        int s = st * 16 + v + 8 * hh;
        int d = t * 16 + ln16;
        size_t addr = ((size_t)(b * S_ + s)) * A_ + h * DH_ + d;
        out_bf[addr] = f2bf(acc[t][v] * inv[v]);
      }
    }
  }
}

// ---------------- convergence freeze: xi = where(diff>eps, xi_new, xi) ----------------
__global__ void freeze_kernel(const float* __restrict__ diff,
                              const float* __restrict__ xi_new,
                              float* __restrict__ xi,
                              unsigned short* __restrict__ xi_bf) {
  int bh = blockIdx.y;
  if (!(sqrtf(diff[bh]) > CONV_EPS_)) return;    // frozen head: keep old xi
  int b = bh >> 2, h = bh & 3;
  int e = blockIdx.x * blockDim.x + threadIdx.x; // 0 .. S_*DH_-1
  int s = e >> 8, d = e & 255;
  size_t addr = ((size_t)(b * S_ + s)) * A_ + h * DH_ + d;
  float v = xi_new[addr];
  xi[addr] = v;
  xi_bf[addr] = f2bf(v);
}

__global__ void zero_diff_kernel(float* __restrict__ diff) {
  if (threadIdx.x < B_ * H_) diff[threadIdx.x] = 0.f;
}

// ---------------- host orchestration ----------------
extern "C" void kernel_launch(void* const* d_in, const int* in_sizes, int n_in,
                              void* d_out, int out_size, void* d_ws, size_t ws_size,
                              hipStream_t stream) {
  const float* hidden   = (const float*)d_in[0];
  const float* memory   = (const float*)d_in[1];
  const float* Wq       = (const float*)d_in[2];
  const float* bq       = (const float*)d_in[3];
  const float* Wk       = (const float*)d_in[4];
  const float* bk       = (const float*)d_in[5];
  const float* Wv       = (const float*)d_in[6];
  const float* bv       = (const float*)d_in[7];
  const float* Wo       = (const float*)d_in[8];
  const float* bo       = (const float*)d_in[9];
  const float* g_stored = (const float*)d_in[10];
  const float* b_stored = (const float*)d_in[11];
  const float* g_state  = (const float*)d_in[12];
  const float* b_state  = (const float*)d_in[13];
  const float* g_proj   = (const float*)d_in[14];
  const float* b_proj   = (const float*)d_in[15];
  float* out = (float*)d_out;

  // bump allocator over d_ws
  char* w = (char*)d_ws;
  auto alloc = [&](size_t bytes) -> void* {
    void* p = (void*)w;
    w += (bytes + 255) & ~(size_t)255;
    return p;
  };
  unsigned short* state_bf  = (unsigned short*)alloc((size_t)B_ * S_ * D_ * 2);
  unsigned short* stored_bf = (unsigned short*)alloc((size_t)N_ * M_ * 2);
  unsigned short* proj_bf   = (unsigned short*)alloc((size_t)N_ * M_ * 2);
  unsigned short* wq_bf     = (unsigned short*)alloc((size_t)A_ * D_ * 2);
  unsigned short* wk_bf     = (unsigned short*)alloc((size_t)A_ * M_ * 2);
  unsigned short* wv_bf     = (unsigned short*)alloc((size_t)A_ * M_ * 2);
  unsigned short* wo_bf     = (unsigned short*)alloc((size_t)D_ * A_ * 2);
  float*          xi_f32    = (float*)alloc((size_t)B_ * S_ * A_ * 4);
  float*          xi_new    = (float*)alloc((size_t)B_ * S_ * A_ * 4);
  unsigned short* xi_bf     = (unsigned short*)alloc((size_t)B_ * S_ * A_ * 2);
  float*          kv_f32    = (float*)alloc((size_t)N_ * A_ * 4);     // reused for k then v
  unsigned short* k_bf      = (unsigned short*)alloc((size_t)N_ * A_ * 2);
  unsigned short* kT_bf     = (unsigned short*)alloc((size_t)A_ * N_ * 2);
  unsigned short* vT_bf     = (unsigned short*)alloc((size_t)A_ * N_ * 2);
  unsigned short* oassoc_bf = (unsigned short*)alloc((size_t)B_ * S_ * A_ * 2);
  float*          diff      = (float*)alloc((size_t)B_ * H_ * 4);

  // 1) LayerNorms -> bf16
  ln_bf16_kernel<<<B_ * S_, 256, 0, stream>>>(hidden, g_state,  b_state,  state_bf,  D_);
  ln_bf16_kernel<<<N_,      256, 0, stream>>>(memory, g_stored, b_stored, stored_bf, M_);
  ln_bf16_kernel<<<N_,      256, 0, stream>>>(memory, g_proj,   b_proj,   proj_bf,   M_);

  // 2) weights -> bf16
  auto cvt = [&](const float* src, unsigned short* dst, size_t n) {
    if (n == 0) return;
    f32_to_bf16_kernel<<<(unsigned)((n + 255) / 256), 256, 0, stream>>>(src, dst, n);
  };
  cvt(Wq, wq_bf, (size_t)A_ * D_);
  cvt(Wk, wk_bf, (size_t)A_ * M_);
  cvt(Wv, wv_bf, (size_t)A_ * M_);
  cvt(Wo, wo_bf, (size_t)D_ * A_);

  // 3) q = state_n @ Wq^T + bq  -> xi (fp32) and xi_bf16
  gemm_bf16_kernel<<<dim3(A_ / 32, (B_ * S_) / 32), 32, 0, stream>>>(
      state_bf, wq_bf, bq, xi_f32, B_ * S_, D_, A_);
  cvt(xi_f32, xi_bf, (size_t)B_ * S_ * A_);

  // 4) k = stored_n @ Wk^T + bk  -> k_bf (N,A) and kT_bf (A,N)
  gemm_bf16_kernel<<<dim3(A_ / 32, N_ / 32), 32, 0, stream>>>(
      stored_bf, wk_bf, bk, kv_f32, N_, M_, A_);
  cvt(kv_f32, k_bf, (size_t)N_ * A_);
  transpose_bf16_kernel<<<(unsigned)(((size_t)N_ * A_ + 255) / 256), 256, 0, stream>>>(
      kv_f32, kT_bf, N_, A_);

  // 5) v = proj_n @ Wv^T + bv -> vT_bf (A,N)   (reuses kv_f32)
  gemm_bf16_kernel<<<dim3(A_ / 32, N_ / 32), 32, 0, stream>>>(
      proj_bf, wv_bf, bv, kv_f32, N_, M_, A_);
  transpose_bf16_kernel<<<(unsigned)(((size_t)N_ * A_ + 255) / 256), 256, 0, stream>>>(
      kv_f32, vT_bf, N_, A_);

  // 6) iterative Hopfield retrieval with per-(b,h) convergence freeze
  dim3 agrid(S_ / 64, H_, B_);   // 4 query tiles per block (one per wave)
  for (int step = 0; step < 3; ++step) {
    zero_diff_kernel<<<1, 32, 0, stream>>>(diff);
    hopfield_attn_kernel<<<agrid, 128, 0, stream>>>(
        xi_bf, k_bf, kT_bf, xi_f32, xi_new, diff, nullptr, /*is_final=*/0);
    freeze_kernel<<<dim3((S_ * DH_) / 256, B_ * H_), 256, 0, stream>>>(
        diff, xi_new, xi_f32, xi_bf);
  }

  // 7) final readout: softmax(beta * xi k^T) @ v  -> bf16 association output
  hopfield_attn_kernel<<<agrid, 128, 0, stream>>>(
      xi_bf, k_bf, vT_bf, xi_f32, nullptr, nullptr, oassoc_bf, /*is_final=*/1);

  // 8) out = assoc @ Wo^T + bo  (fp32 result)
  gemm_bf16_kernel<<<dim3(D_ / 32, (B_ * S_) / 32), 32, 0, stream>>>(
      oassoc_bf, wo_bf, bo, out, B_ * S_, A_, D_);
}
